// GeoFormerMix_46643344835056
// MI455X (gfx1250) — matrix operand order repaired and checked by reading, not compiled, not run on hardware
//
#include <hip/hip_runtime.h>

#define Nn    10000
#define Ee    160000
#define INC   128
#define HID   256
#define OUTC  64
#define LNUM  2
#define HNUM  8
#define DH    32
#define FF    1024
#define SCALEA 0.17677669529663687f /* 1/sqrt(32) */

typedef __attribute__((ext_vector_type(16))) _Float16 v16h;
typedef __attribute__((ext_vector_type(8)))  _Float16 v8h;
typedef __attribute__((ext_vector_type(8)))  float    v8f;

#define BK  32
#define PAD 8

__device__ __forceinline__ v16h frag16(const _Float16* p0, const _Float16* p1) {
  v8h lo = *reinterpret_cast<const v8h*>(p0);
  v8h hi = *reinterpret_cast<const v8h*>(p1);
  return __builtin_shufflevector(lo, hi,
      0, 1, 2, 3, 4, 5, 6, 7, 8, 9, 10, 11, 12, 13, 14, 15);
}

// ---------------------------------------------------------------------------
// GEMM: C[M x Nout] = A[M x K] @ W[Nout x K]^T + bias, f16 WMMA, f32 accum.
// Block tile 128 x (32*NI); 8 waves as 4(M) x 2(N); each wave owns a
// 32 x (16*NI) register tile = 2 x NI WMMA accumulators (4 WMMAs/iter for NI=2).
// epilogue: 0 = none, 1 = exact GELU
// ---------------------------------------------------------------------------
template <int NI>
__global__ __launch_bounds__(256) void gemm_wmma_t(
    const float* __restrict__ A, int lda,
    const float* __restrict__ W, int K,
    const float* __restrict__ bias,
    float* __restrict__ C, int ldc,
    int M, int epilogue)
{
  constexpr int TBM = 128;
  constexpr int TBN = 32 * NI;
  __shared__ __align__(16) _Float16 sA[TBM][BK + PAD];
  __shared__ __align__(16) _Float16 sB[TBN][BK + PAD];

  const int tid  = threadIdx.x;
  const int lane = tid & 31;
  const int wave = tid >> 5;
  const int wm   = wave >> 1;   // 0..3  (32 rows each)
  const int wn   = wave & 1;    // 0..1  (16*NI cols each)
  const int m0   = blockIdx.x * TBM;
  const int n0   = blockIdx.y * TBN;

  v8f acc[2][NI];
#pragma unroll
  for (int i = 0; i < 2; ++i)
#pragma unroll
    for (int j = 0; j < NI; ++j)
      acc[i][j] = (v8f){0.f, 0.f, 0.f, 0.f, 0.f, 0.f, 0.f, 0.f};

  const int arow = tid >> 1;        // 0..127
  const int acg  = (tid & 1) << 4;  // 0 or 16
  const int brow = tid >> 2;        // 0..63
  const int bcg  = (tid & 3) << 3;  // 0,8,16,24

  for (int k0 = 0; k0 < K; k0 += BK) {
    // ---- stage A tile (TBM x 32 f32 -> f16), bounds-checked on M
    {
      const int gm = m0 + arow;
      float4 u0 = make_float4(0.f, 0.f, 0.f, 0.f), u1 = u0, u2 = u0, u3 = u0;
      if (gm < M) {
        const float4* p = reinterpret_cast<const float4*>(A + (size_t)gm * lda + k0 + acg);
        u0 = p[0]; u1 = p[1]; u2 = p[2]; u3 = p[3];
      }
      _Float16* d = &sA[arow][acg];
      d[0]  = (_Float16)u0.x; d[1]  = (_Float16)u0.y;
      d[2]  = (_Float16)u0.z; d[3]  = (_Float16)u0.w;
      d[4]  = (_Float16)u1.x; d[5]  = (_Float16)u1.y;
      d[6]  = (_Float16)u1.z; d[7]  = (_Float16)u1.w;
      d[8]  = (_Float16)u2.x; d[9]  = (_Float16)u2.y;
      d[10] = (_Float16)u2.z; d[11] = (_Float16)u2.w;
      d[12] = (_Float16)u3.x; d[13] = (_Float16)u3.y;
      d[14] = (_Float16)u3.z; d[15] = (_Float16)u3.w;
    }
    // ---- stage W tile (TBN x 32 f32 -> f16); Nout is a multiple of TBN
    if (brow < TBN) {
      const int gn = n0 + brow;
      const float4* p = reinterpret_cast<const float4*>(W + (size_t)gn * K + k0 + bcg);
      float4 u0 = p[0], u1 = p[1];
      _Float16* d = &sB[brow][bcg];
      d[0] = (_Float16)u0.x; d[1] = (_Float16)u0.y;
      d[2] = (_Float16)u0.z; d[3] = (_Float16)u0.w;
      d[4] = (_Float16)u1.x; d[5] = (_Float16)u1.y;
      d[6] = (_Float16)u1.z; d[7] = (_Float16)u1.w;
    }
    __syncthreads();

    // ---- fragments (ISA 7.12.2 layouts)
    const int kbA = (lane >> 4) << 3;  // A: kbase 0/8, elems 0..7 -> kb.., 8..15 -> kb+16..
    const int kbB = (lane >> 4) << 3;  // B: kbase 0/8, elems 0..7 -> kb.., 8..15 -> kb+16..
    v16h afrag[2], bfrag[NI];
#pragma unroll
    for (int i = 0; i < 2; ++i) {
      const int ml = wm * 32 + i * 16 + (lane & 15);
      afrag[i] = frag16(&sA[ml][kbA], &sA[ml][kbA + 16]);
    }
#pragma unroll
    for (int j = 0; j < NI; ++j) {
      const int nl = wn * (16 * NI) + j * 16 + (lane & 15);
      bfrag[j] = frag16(&sB[nl][kbB], &sB[nl][kbB + 16]);
    }
#pragma unroll
    for (int i = 0; i < 2; ++i)
#pragma unroll
      for (int j = 0; j < NI; ++j)
        acc[i][j] = __builtin_amdgcn_wmma_f32_16x16x32_f16(
            false, afrag[i], false, bfrag[j], (short)0, acc[i][j], false, false);
    __syncthreads();
  }

  // ---- epilogue: C/D layout: VGPR r -> M = r + (lane>=16 ? 8 : 0), N = lane&15
#pragma unroll
  for (int i = 0; i < 2; ++i) {
#pragma unroll
    for (int j = 0; j < NI; ++j) {
      const int gn = n0 + wn * (16 * NI) + j * 16 + (lane & 15);
      const int mb = m0 + wm * 32 + i * 16 + ((lane >> 4) << 3);
      const float bv = bias[gn];
#pragma unroll
      for (int r = 0; r < 8; ++r) {
        const int gm = mb + r;
        if (gm < M) {
          float xv = acc[i][j][r] + bv;
          if (epilogue == 1)
            xv = 0.5f * xv * (1.0f + erff(xv * 0.70710678118654752f));
          C[(size_t)gm * ldc + gn] = xv;
        }
      }
    }
  }
}

// ---------------------------------------------------------------------------
// helpers
// ---------------------------------------------------------------------------
__device__ __forceinline__ float waveRed(float v) {
#pragma unroll
  for (int m = 16; m; m >>= 1) v += __shfl_xor(v, m, 32);
  return v;
}

__device__ __forceinline__ void atomicMaxF(float* addr, float val) {
  unsigned int* ua = reinterpret_cast<unsigned int*>(addr);
  unsigned int old = __float_as_uint(*addr);
  while (__uint_as_float(old) < val) {
    unsigned int assumed = old;
    old = atomicCAS(ua, assumed, __float_as_uint(val));
    if (old == assumed) break;
  }
}

__global__ void k_fill(float* __restrict__ p, float v, int n) {
  const int i = blockIdx.x * blockDim.x + threadIdx.x;
  if (i < n) p[i] = v;
}

// ---------------------------------------------------------------------------
// Poincare log0 and sphere log0 of h (one block of 256 per row)
// ---------------------------------------------------------------------------
__global__ __launch_bounds__(HID) void k_logmaps(
    const float* __restrict__ h, float* __restrict__ lp, float* __restrict__ ls)
{
  __shared__ float red[HID];
  const int row = blockIdx.x;
  const int t = threadIdx.x;
  const size_t idx = (size_t)row * HID + t;
  const float x = h[idx];
  red[t] = x * x;
  __syncthreads();
  for (int o = HID / 2; o; o >>= 1) { if (t < o) red[t] += red[t + o]; __syncthreads(); }
  const float s2 = red[0];
  const float n  = fmaxf(sqrtf(s2), 1e-15f);
  // Poincare: atanh(min(n, 1-1e-5)) * x / n
  const float nc = fminf(n, 1.0f - 1e-5f);
  lp[idx] = atanhf(nc) * x / n;
  // Sphere: project to unit, theta = acos(clip(x_last)), out = theta*u/||u||
  const float xl = h[(size_t)row * HID + (HID - 1)] / n;
  const float ct = fminf(fmaxf(xl, -1.0f + 1e-7f), 1.0f - 1e-7f);
  const float theta = acosf(ct);
  const float xn = x / n;
  const float sn2 = s2 / (n * n);
  const float nu = fmaxf(sqrtf(fmaxf(sn2 - xl * xl, 0.0f)), 1e-15f);
  const float u = (t == HID - 1) ? 0.0f : xn;
  ls[idx] = theta * u / nu;
}

// ---------------------------------------------------------------------------
// exp0 + proj maps, wave-per-row (DH == wave32)
// ---------------------------------------------------------------------------
__global__ void k_poincare_exp(float* __restrict__ z, int nrows) {
  const int row = (blockIdx.x * blockDim.x + threadIdx.x) >> 5;
  const int lane = threadIdx.x & 31;
  if (row >= nrows) return;
  const size_t idx = (size_t)row * DH + lane;
  const float v = z[idx];
  const float s = waveRed(v * v);
  const float n = fmaxf(sqrtf(s), 1e-15f);
  float o = tanhf(n) * v / n;
  const float s2 = waveRed(o * o);
  const float n2 = fmaxf(sqrtf(s2), 1e-15f);
  const float maxn = 1.0f - 1e-5f;
  if (n2 > maxn) o = o / n2 * maxn;
  z[idx] = o;
}

__global__ void k_sphere_exp(float* __restrict__ z, int nrows) {
  const int row = (blockIdx.x * blockDim.x + threadIdx.x) >> 5;
  const int lane = threadIdx.x & 31;
  if (row >= nrows) return;
  const size_t idx = (size_t)row * DH + lane;
  const float v = z[idx];
  const float vt = (lane == DH - 1) ? 0.0f : v;
  const float s = waveRed(vt * vt);
  const float n = fmaxf(sqrtf(s), 1e-15f);
  float o = (lane == DH - 1) ? cosf(n) : (sinf(n) / n) * v;
  const float s2 = waveRed(o * o);
  const float nn = fmaxf(sqrtf(s2), 1e-15f);
  z[idx] = o / nn;
}

// ---------------------------------------------------------------------------
// edge phase: alpha + segment max, exp + segment sum, normalized scatter
// ---------------------------------------------------------------------------
__global__ void k_edge_alpha(
    const float* __restrict__ q, const float* __restrict__ k,
    const int* __restrict__ row, const int* __restrict__ col,
    float* __restrict__ alphaE, float* __restrict__ mrow, int curv)
{
  const int e = (blockIdx.x * blockDim.x + threadIdx.x) >> 5;
  const int lane = threadIdx.x & 31;
  if (e >= Ee) return;
  const int r = row[e], c = col[e];
  const float qv = q[(size_t)r * DH + lane];
  const float kv = k[(size_t)c * DH + lane];
  const float dot = waveRed(qv * kv);
  const float q2  = waveRed(qv * qv);
  const float k2  = waveRed(kv * kv);
  const float dd  = waveRed((qv - kv) * (qv - kv));
  if (lane == 0) {
    float a = dot * SCALEA;
    if (curv < 0) {
      const float den = fmaxf((1.0f - q2) * (1.0f - k2), 1e-15f);
      const float arg = fmaxf(1.0f + 2.0f * dd / den, 1.0f + 1e-7f);
      a -= 0.1f * acoshf(arg);
    } else if (curv > 0) {
      const float cc = fminf(fmaxf(dot, -1.0f + 1e-7f), 1.0f - 1e-7f);
      a -= 0.1f * acosf(cc);
    }
    alphaE[e] = a;
    atomicMaxF(&mrow[r], a);
  }
}

__global__ void k_edge_exp(
    const int* __restrict__ row, float* __restrict__ alphaE,
    const float* __restrict__ mrow, float* __restrict__ srow)
{
  const int e = blockIdx.x * blockDim.x + threadIdx.x;
  if (e >= Ee) return;
  const int r = row[e];
  const float a = expf(alphaE[e] - mrow[r]);
  alphaE[e] = a;
  atomicAdd(&srow[r], a);
}

__global__ void k_edge_scatter(
    const float* __restrict__ v,
    const int* __restrict__ row, const int* __restrict__ col,
    const float* __restrict__ alphaE, const float* __restrict__ srow,
    float* __restrict__ attpre, int hoff)
{
  const int e = (blockIdx.x * blockDim.x + threadIdx.x) >> 5;
  const int lane = threadIdx.x & 31;
  if (e >= Ee) return;
  const int r = row[e], c = col[e];
  const float w = alphaE[e] / (srow[r] + 1e-16f);
  atomicAdd(&attpre[(size_t)r * HID + hoff + lane], v[(size_t)c * DH + lane] * w);
}

// ---------------------------------------------------------------------------
// out = LayerNorm(a + b) * g + beta  (one block of 256 per row)
// ---------------------------------------------------------------------------
__global__ __launch_bounds__(HID) void k_add_ln(
    const float* __restrict__ a, const float* __restrict__ b,
    const float* __restrict__ g, const float* __restrict__ bet,
    float* __restrict__ outp)
{
  __shared__ float red[HID];
  const int row = blockIdx.x, t = threadIdx.x;
  const size_t idx = (size_t)row * HID + t;
  const float x = a[idx] + b[idx];
  red[t] = x; __syncthreads();
  for (int o = HID / 2; o; o >>= 1) { if (t < o) red[t] += red[t + o]; __syncthreads(); }
  const float mean = red[0] * (1.0f / HID);
  __syncthreads();
  const float d = x - mean;
  red[t] = d * d; __syncthreads();
  for (int o = HID / 2; o; o >>= 1) { if (t < o) red[t] += red[t + o]; __syncthreads(); }
  const float var = red[0] * (1.0f / HID);
  outp[idx] = d * rsqrtf(var + 1e-5f) * g[t] + bet[t];
}

// ---------------------------------------------------------------------------
static inline int cdiv(int a, int b) { return (a + b - 1) / b; }

extern "C" void kernel_launch(void* const* d_in, const int* in_sizes, int n_in,
                              void* d_out, int out_size, void* d_ws, size_t ws_size,
                              hipStream_t stream) {
  const float* x    = (const float*)d_in[0];
  const int*   ei   = (const int*)d_in[1];
  const float* embW = (const float*)d_in[2];
  const float* embb = (const float*)d_in[3];
  const float* qW   = (const float*)d_in[4];
  const float* qb   = (const float*)d_in[5];
  const float* kW   = (const float*)d_in[6];
  const float* kb   = (const float*)d_in[7];
  const float* vW   = (const float*)d_in[8];
  const float* vb   = (const float*)d_in[9];
  const float* loW  = (const float*)d_in[10];
  const float* lob  = (const float*)d_in[11];
  const float* f1W  = (const float*)d_in[12];
  const float* f1b  = (const float*)d_in[13];
  const float* f2W  = (const float*)d_in[14];
  const float* f2b  = (const float*)d_in[15];
  const float* n1g  = (const float*)d_in[16];
  const float* n1b  = (const float*)d_in[17];
  const float* n2g  = (const float*)d_in[18];
  const float* n2b  = (const float*)d_in[19];
  const float* outW = (const float*)d_in[20];
  const float* outb = (const float*)d_in[21];
  const int* erow = ei;
  const int* ecol = ei + Ee;

  float* ws = (float*)d_ws;
  size_t off = 0;
  auto take = [&](size_t n) { float* p = ws + off; off += n; return p; };
  float* h      = take((size_t)Nn * HID);
  float* lp     = take((size_t)Nn * HID);
  float* ls     = take((size_t)Nn * HID);
  float* attpre = take((size_t)Nn * HID);
  float* att    = take((size_t)Nn * HID);
  float* h1     = take((size_t)Nn * HID);
  float* ff2    = take((size_t)Nn * HID);
  float* ff1    = take((size_t)Nn * FF);
  float* qbuf   = take((size_t)Nn * DH);
  float* kbuf   = take((size_t)Nn * DH);
  float* vbuf   = take((size_t)Nn * DH);
  float* alphaE = take((size_t)Ee);
  float* mrow   = take((size_t)Nn);
  float* srow   = take((size_t)Nn);

  const dim3 blk(256);
  const int curvs[HNUM] = {-1, -1, 0, 0, 1, 1, 1, 1};

  auto gemm = [&](const float* A, int lda, const float* W, int K,
                  const float* bias, float* C, int ldc, int M, int Nout, int epi) {
    if (Nout % 64 == 0) {
      dim3 grid(cdiv(M, 128), Nout / 64);
      gemm_wmma_t<2><<<grid, blk, 0, stream>>>(A, lda, W, K, bias, C, ldc, M, epi);
    } else {
      dim3 grid(cdiv(M, 128), Nout / 32);
      gemm_wmma_t<1><<<grid, blk, 0, stream>>>(A, lda, W, K, bias, C, ldc, M, epi);
    }
  };

  // h = x @ emb_W^T + emb_b
  gemm(x, INC, embW, INC, embb, h, HID, Nn, HID, 0);

  const int nodeWaveBlocks = cdiv(Nn * 32, 256);
  const int edgeWaveBlocks = cdiv(Ee * 32, 256);
  const int edgeBlocks     = cdiv(Ee, 256);

  for (int l = 0; l < LNUM; ++l) {
    k_logmaps<<<dim3(Nn), dim3(HID), 0, stream>>>(h, lp, ls);
    k_fill<<<dim3(cdiv(Nn * HID, 256)), blk, 0, stream>>>(attpre, 0.0f, Nn * HID);

    for (int hh = 0; hh < HNUM; ++hh) {
      const int cv = curvs[hh];
      const float* Ain = (cv < 0) ? lp : (cv == 0 ? h : ls);
      const size_t woff = ((size_t)l * HNUM + hh) * (size_t)DH * HID;
      const size_t boff = ((size_t)l * HNUM + hh) * DH;
      gemm(Ain, HID, qW + woff, HID, qb + boff, qbuf, DH, Nn, DH, 0);
      gemm(Ain, HID, kW + woff, HID, kb + boff, kbuf, DH, Nn, DH, 0);
      gemm(Ain, HID, vW + woff, HID, vb + boff, vbuf, DH, Nn, DH, 0);
      if (cv < 0) {
        k_poincare_exp<<<dim3(nodeWaveBlocks), blk, 0, stream>>>(qbuf, Nn);
        k_poincare_exp<<<dim3(nodeWaveBlocks), blk, 0, stream>>>(kbuf, Nn);
        k_poincare_exp<<<dim3(nodeWaveBlocks), blk, 0, stream>>>(vbuf, Nn);
      } else if (cv > 0) {
        k_sphere_exp<<<dim3(nodeWaveBlocks), blk, 0, stream>>>(qbuf, Nn);
        k_sphere_exp<<<dim3(nodeWaveBlocks), blk, 0, stream>>>(kbuf, Nn);
        k_sphere_exp<<<dim3(nodeWaveBlocks), blk, 0, stream>>>(vbuf, Nn);
      }
      k_fill<<<dim3(cdiv(Nn, 256)), blk, 0, stream>>>(mrow, -3.0e38f, Nn);
      k_fill<<<dim3(cdiv(Nn, 256)), blk, 0, stream>>>(srow, 0.0f, Nn);
      k_edge_alpha<<<dim3(edgeWaveBlocks), blk, 0, stream>>>(
          qbuf, kbuf, erow, ecol, alphaE, mrow, cv);
      k_edge_exp<<<dim3(edgeBlocks), blk, 0, stream>>>(erow, alphaE, mrow, srow);
      k_edge_scatter<<<dim3(edgeWaveBlocks), blk, 0, stream>>>(
          vbuf, erow, ecol, alphaE, srow, attpre, hh * DH);
    }

    // att = attpre @ loW^T + lob ; h1 = LN(h + att)
    gemm(attpre, HID, loW + (size_t)l * HID * HID, HID, lob + (size_t)l * HID,
         att, HID, Nn, HID, 0);
    k_add_ln<<<dim3(Nn), dim3(HID), 0, stream>>>(h, att, n1g + l * HID, n1b + l * HID, h1);

    // ff1 = gelu(h1 @ f1W^T + f1b) ; ff2 = ff1 @ f2W^T + f2b ; h = LN(h1 + ff2)
    gemm(h1, HID, f1W + (size_t)l * FF * HID, HID, f1b + (size_t)l * FF,
         ff1, FF, Nn, FF, 1);
    gemm(ff1, FF, f2W + (size_t)l * HID * FF, FF, f2b + (size_t)l * HID,
         ff2, HID, Nn, HID, 0);
    k_add_ln<<<dim3(Nn), dim3(HID), 0, stream>>>(h1, ff2, n2g + l * HID, n2b + l * HID, h);
  }

  // out = h @ out_W^T + out_b
  gemm(h, HID, outW, HID, outb, (float*)d_out, OUTC, Nn, OUTC, 0);
}
